// AttentionHead_7232724927256
// MI455X (gfx1250) — matrix operand order repaired and checked by reading, not compiled, not run on hardware
//
#include <hip/hip_runtime.h>
#include <hip/hip_fp16.h>
#include <math.h>
#include <stdint.h>

typedef __attribute__((ext_vector_type(16))) _Float16 v16h;
typedef __attribute__((ext_vector_type(8)))  float    v8f;

#define DIN   1024
#define HID   128
#define SEQ   4096
#define NROWS 16384               // 4 batches * 4096
#define ATTN_SCALE 0.08838834764831845f   // 1/sqrt(128)

union HFrag { float4 f4[2]; v16h h; };
union H8    { float4 f4;    _Float16 h[8]; };

// lds byte offset of a __shared__ object (LDS aperture base has zero low bits)
static __device__ inline unsigned lds_off(const void* p) {
  return (unsigned)(uintptr_t)p;
}
// gfx1250 async copy: 16B global -> LDS, tracked by ASYNCcnt
static __device__ inline void async_ld_b128(unsigned dst_lds, const void* src) {
  asm volatile("global_load_async_to_lds_b128 %0, %1, off"
               :: "v"(dst_lds), "v"((unsigned long long)(uintptr_t)src)
               : "memory");
}

// reduce across the 16 lanes of each half-wave (xor masks < 16 never cross halves)
static __device__ inline float hmax16(float v) {
#pragma unroll
  for (int m = 8; m >= 1; m >>= 1) v = fmaxf(v, __shfl_xor(v, m, 32));
  return v;
}
static __device__ inline float hsum16(float v) {
#pragma unroll
  for (int m = 8; m >= 1; m >>= 1) v += __shfl_xor(v, m, 32);
  return v;
}

// ---------------------------------------------------------------------------
// Kernel 1: W [Din][H] fp32  ->  Wt [mat][H][Din] f16 (transposed, contiguous K)
// ---------------------------------------------------------------------------
__global__ void prep_weights(const float* __restrict__ Wq,
                             const float* __restrict__ Wk,
                             const float* __restrict__ Wv,
                             _Float16* __restrict__ Wt) {
  int idx = blockIdx.x * blockDim.x + threadIdx.x;
  if (idx >= 3 * HID * DIN) return;
  int mat = idx / (HID * DIN);
  int rem = idx % (HID * DIN);
  int h = rem / DIN;
  int k = rem % DIN;
  const float* W = (mat == 0) ? Wq : ((mat == 1) ? Wk : Wv);
  Wt[idx] = (_Float16)W[k * HID + h];
}

// ---------------------------------------------------------------------------
// Kernel 2: fused QKV projection. One block per 16-row x tile: convert the
// tile fp32->f16 into LDS once, then every wave (8 n-tiles) runs 3 WMMAs per
// K-step (Q,K,V share the A fragment). x is read exactly once from global.
// V is stored transposed AND 32-key-slot-interleaved (s = 2*(p&15) + (p>>4))
// so flash_attn's packed-P trick works with contiguous V reads.
// grid = 1024, block = 256 (8 waves)
// ---------------------------------------------------------------------------
__global__ void qkv_proj(const float* __restrict__ x,
                         const _Float16* __restrict__ Wt,
                         const float* __restrict__ bq,
                         const float* __restrict__ bk,
                         const float* __restrict__ bv,
                         _Float16* __restrict__ Qh,
                         _Float16* __restrict__ Kh,
                         _Float16* __restrict__ Vt) {
  __shared__ _Float16 Axh[16][DIN + 8];   // +8 halves pad: stride 2064B (+4 banks/row)

  const int mtile = blockIdx.x;
  const int tid   = threadIdx.x;
  const int nt    = tid >> 5;
  const int lane  = tid & 31;
  const int lm = lane & 15, hs = lane >> 4;

  // ---- cooperative fp32 -> f16 conversion of the 16x1024 A tile into LDS
#pragma unroll
  for (int t = 0; t < 8; ++t) {
    const int c    = tid + t * 256;        // 0..2047 chunks of 8 halves
    const int row  = c >> 7;
    const int koff = (c & 127) * 8;
    const float* src = x + (long)(mtile * 16 + row) * DIN + koff;
    const float4 f0 = ((const float4*)src)[0];
    const float4 f1 = ((const float4*)src)[1];
    H8 o;
    o.h[0] = (_Float16)f0.x; o.h[1] = (_Float16)f0.y;
    o.h[2] = (_Float16)f0.z; o.h[3] = (_Float16)f0.w;
    o.h[4] = (_Float16)f1.x; o.h[5] = (_Float16)f1.y;
    o.h[6] = (_Float16)f1.z; o.h[7] = (_Float16)f1.w;
    *(float4*)&Axh[row][koff] = o.f4;
  }
  __syncthreads();

  const int ncol = nt * 16 + lm;           // B column for this lane
  const _Float16* Wm0 = Wt + (long)ncol * DIN;

  v8f acc[3];
  const v8f vzero = {};
#pragma unroll
  for (int m = 0; m < 3; ++m) acc[m] = vzero;

#pragma unroll 2
  for (int k0 = 0; k0 < DIN; k0 += 32) {
    HFrag a;   // A 16x32: row lm, 16 contiguous K at k0+hs*16 (ds_load_b128 x2)
    a.f4[0] = *(const float4*)&Axh[lm][k0 + hs * 16];
    a.f4[1] = *(const float4*)&Axh[lm][k0 + hs * 16 + 8];
#pragma unroll
    for (int m = 0; m < 3; ++m) {
      HFrag b; // B 32x16: col ncol, 16 contiguous K at k0+hs*16
      const float4* wp = (const float4*)(Wm0 + (long)m * (HID * DIN) + k0 + hs * 16);
      b.f4[0] = wp[0];
      b.f4[1] = wp[1];
      acc[m] = __builtin_amdgcn_wmma_f32_16x16x32_f16(false, a.h, false, b.h,
                                                      (short)0, acc[m], false, false);
    }
  }

  const float bval[3] = { bq[ncol], bk[ncol], bv[ncol] };
#pragma unroll
  for (int r = 0; r < 8; ++r) {
    const int orow = mtile * 16 + r + hs * 8;       // C layout row
    Qh[(long)orow * HID + ncol] = (_Float16)(acc[0][r] + bval[0]);
    Kh[(long)orow * HID + ncol] = (_Float16)(acc[1][r] + bval[1]);
    // transposed V with 32-row-group slot interleave
    const int p = orow & 31;
    const int s = ((p & 15) << 1) | (p >> 4);
    Vt[(long)ncol * NROWS + (orow & ~31) + s] = (_Float16)(acc[2][r] + bval[2]);
  }
}

// ---------------------------------------------------------------------------
// Kernel 3: flash attention, double-buffered async K/V staging in LDS.
// Block = 4 waves = 64 consecutive queries (grid 256 for wider WGP coverage).
// Per 32-key tile: 128 threads issue 8x global_load_async_to_lds_b128 for the
// NEXT tile, s_wait_asynccnt retires the current one, s_barrier publishes it,
// then each wave: 8 score WMMAs + online softmax + packed-P LDS transpose +
// 8 P*V WMMAs. Key slots inside a tile are interleaved (matching Vt) so
// (p0,p1) pack into a single v_cvt_pk + ds_store_b32.
// ---------------------------------------------------------------------------
struct FlashSmem {
  _Float16 K[2][32][136];   // 32 keys x 128 h, +8 pad  (row 272B -> +4 banks)
  _Float16 V[2][128][40];   // 128 h x 32 slots, +8 pad (row  80B -> +20 banks)
  _Float16 P[4][16][40];    // per-wave prob tile, 40-half stride (halves split banks)
};

__global__ void flash_attn(const _Float16* __restrict__ Qh,
                           const _Float16* __restrict__ Kh,
                           const _Float16* __restrict__ Vt,
                           float* __restrict__ out) {
  __shared__ FlashSmem sm;

  const int tid  = threadIdx.x;             // 0..127
  const int w    = tid >> 5;                // 0..3
  const int lane = tid & 31;
  const int lm = lane & 15, hs = lane >> 4;
  const int g0    = (blockIdx.x * 4 + w) * 16;     // global query row base
  const int bbase = (blockIdx.x * 64) & ~(SEQ - 1);

  auto issue_tile = [&](int kt, int b) {
    const _Float16* kg = Kh + (long)(bbase + kt) * HID;
#pragma unroll
    for (int t = 0; t < 4; ++t) {           // 512 K chunks / 128 threads
      const int c = tid + t * 128;
      const int row = c >> 4, off = (c & 15) * 8;
      async_ld_b128(lds_off(&sm.K[b][row][off]), kg + (long)row * HID + off);
    }
    const _Float16* vg = Vt + bbase + kt;   // already slot-interleaved
#pragma unroll
    for (int t = 0; t < 4; ++t) {           // 512 V chunks / 128 threads
      const int c = tid + t * 128;
      const int row = c >> 2, off = (c & 3) * 8;
      async_ld_b128(lds_off(&sm.V[b][row][off]), vg + (long)row * NROWS + off);
    }
  };

  // ---- Q tile in A-layout: whole 16x128 tile in registers
  HFrag q[4];
  const _Float16* qrow = Qh + (long)(g0 + lm) * HID;
#pragma unroll
  for (int kc = 0; kc < 4; ++kc) {
    q[kc].f4[0] = *(const float4*)(qrow + kc * 32 + hs * 8);
    q[kc].f4[1] = *(const float4*)(qrow + kc * 32 + hs * 8 + 16);
  }

  float mrow[8], lrow[8];
  v8f O[8];
  const v8f vzero = {};
#pragma unroll
  for (int r = 0; r < 8; ++r) { mrow[r] = -INFINITY; lrow[r] = 0.0f; }
#pragma unroll
  for (int t = 0; t < 8; ++t) O[t] = vzero;

  issue_tile(0, 0);

  const int NT = SEQ / 32;                  // 128 key tiles
  for (int it = 0; it < NT; ++it) {
    const int cur = it & 1;
    if (it + 1 < NT) {
      issue_tile((it + 1) * 32, cur ^ 1);
      asm volatile("s_wait_asynccnt 0x8" ::: "memory");  // current tile done
    } else {
      asm volatile("s_wait_asynccnt 0x0" ::: "memory");
    }
    __syncthreads();                        // tile visible to all waves

    // ---- scores: two 16x16 tiles from LDS K (phys key order), fp32 acc
    v8f s[2];
#pragma unroll
    for (int j = 0; j < 2; ++j) {
      v8f a2 = vzero;
      const _Float16* krow = &sm.K[cur][j * 16 + lm][0];
#pragma unroll
      for (int kc = 0; kc < 4; ++kc) {
        HFrag b;   // B 32x16: col = key (lm), h at hs*8 and +16
        b.f4[0] = *(const float4*)(krow + kc * 32 + hs * 8);
        b.f4[1] = *(const float4*)(krow + kc * 32 + hs * 8 + 16);
        a2 = __builtin_amdgcn_wmma_f32_16x16x32_f16(false, q[kc].h, false, b.h,
                                                    (short)0, a2, false, false);
      }
      s[j] = a2;
    }

    // ---- online softmax over these 32 keys (C-layout row r + hs*8)
#pragma unroll
    for (int r = 0; r < 8; ++r) {
      float s0 = s[0][r] * ATTN_SCALE;   // phys key lm
      float s1 = s[1][r] * ATTN_SCALE;   // phys key lm + 16
      float mt = hmax16(fmaxf(s0, s1));
      float mn = fmaxf(mrow[r], mt);
      float alpha = __expf(mrow[r] - mn);
      mrow[r] = mn;
      float p0 = __expf(s0 - mn);
      float p1 = __expf(s1 - mn);
      lrow[r] = lrow[r] * alpha + hsum16(p0 + p1);
#pragma unroll
      for (int t = 0; t < 8; ++t) O[t][r] *= alpha;
      // packed P: slots (2*lm, 2*lm+1) = phys keys (lm, lm+16)
      union { __fp16 h2 __attribute__((ext_vector_type(2))); unsigned u; } pk;
      pk.h2 = __builtin_amdgcn_cvt_pkrtz(p0, p1);
      *(unsigned*)&sm.P[w][r + hs * 8][2 * lm] = pk.u;
    }

    // ---- read P back in A-layout (slot order matches interleaved Vt)
    HFrag pa;
    pa.f4[0] = *(const float4*)(&sm.P[w][lm][hs * 8]);
    pa.f4[1] = *(const float4*)(&sm.P[w][lm][hs * 8 + 16]);

    // ---- O += P @ V from LDS V (8 h-tiles, K = 32 slots)
#pragma unroll
    for (int t = 0; t < 8; ++t) {
      HFrag b;  // B 32x16: col = h (t*16+lm), slots at hs*8 and +16
      const _Float16* vrow = &sm.V[cur][t * 16 + lm][0];
      b.f4[0] = *(const float4*)(vrow + hs * 8);
      b.f4[1] = *(const float4*)(vrow + hs * 8 + 16);
      O[t] = __builtin_amdgcn_wmma_f32_16x16x32_f16(false, pa.h, false, b.h,
                                                    (short)0, O[t], false, false);
    }
    __syncthreads();   // everyone done reading buf[cur] before it is refilled
  }

  // ---- epilogue: divide by row sums, store fp32 output
#pragma unroll
  for (int r = 0; r < 8; ++r) {
    const float inv = 1.0f / lrow[r];
    const int orow = g0 + r + hs * 8;
#pragma unroll
    for (int t = 0; t < 8; ++t)
      out[(long)orow * HID + t * 16 + lm] = O[t][r] * inv;
  }
}

// ---------------------------------------------------------------------------
extern "C" void kernel_launch(void* const* d_in, const int* in_sizes, int n_in,
                              void* d_out, int out_size, void* d_ws, size_t ws_size,
                              hipStream_t stream) {
  const float* x  = (const float*)d_in[0];
  const float* Wq = (const float*)d_in[1];
  const float* bq = (const float*)d_in[2];
  const float* Wk = (const float*)d_in[3];
  const float* bk = (const float*)d_in[4];
  const float* Wv = (const float*)d_in[5];
  const float* bv = (const float*)d_in[6];
  float* out = (float*)d_out;

  // workspace layout (f16): Wt 768KB | Qh 4MB | Kh 4MB | Vt 4MB  (~12.8 MB)
  char* ws = (char*)d_ws;
  _Float16* Wt = (_Float16*)ws;
  _Float16* Qh = (_Float16*)(ws + 786432);
  _Float16* Kh = (_Float16*)(ws + 786432 + 4194304);
  _Float16* Vt = (_Float16*)(ws + 786432 + 2 * 4194304);

  prep_weights<<<dim3((3 * HID * DIN + 255) / 256), 256, 0, stream>>>(Wq, Wk, Wv, Wt);
  qkv_proj<<<dim3(1024), 256, 0, stream>>>(x, Wt, bq, bk, bv, Qh, Kh, Vt);
  flash_attn<<<dim3(256), 128, 0, stream>>>(Qh, Kh, Vt, out);
}
